// CrossModalAttention_15779709846340
// MI455X (gfx1250) — matrix operand order repaired and checked by reading, not compiled, not run on hardware
//
#include <hip/hip_runtime.h>
#include <hip/hip_bf16.h>
#include <math.h>
#include <stdint.h>

typedef __attribute__((ext_vector_type(16))) _Float16 v16h;
typedef __attribute__((ext_vector_type(8)))  _Float16 v8h;
typedef __attribute__((ext_vector_type(8)))  float    v8f;

#define B_   16
#define TQ_  512
#define TK_  512
#define D_   1024
#define H_   16
#define HD_  64
#define NEGV (-10000.0f)
#define EPS_ 1e-5f

// ---------------------------------------------------------------------------
// WMMA fragment loaders (wave32), this code's contiguous k order k = 16*(l/16) + s for A and B.
// A (16x32 f16, row-major, K contiguous):
//   lane l: m = l&15 ; kb = (l<16)?0:16 ; VGPR0..7 = K[kb..kb+15]
// B (32x16 f16 = 16 rows of W, K contiguous):
//   lane l: n = l&15 ; kb = (l<16)?0:16 ; VGPR0..7 = K[kb..kb+15]
// ---------------------------------------------------------------------------
static __device__ inline v16h ld_a16(const _Float16* __restrict__ A, int lda, int lane) {
  const int m  = lane & 15;
  const int kb = (lane < 16) ? 0 : 16;
  union { v16h v; v8h h[2]; } u;
  u.h[0] = *(const v8h*)(A + (size_t)m * lda + kb);
  u.h[1] = *(const v8h*)(A + (size_t)m * lda + 8 + kb);
  return u.v;
}

static __device__ inline v16h ld_a32(const float* __restrict__ A, int lda, int lane) {
  const int m  = lane & 15;
  const int kb = (lane < 16) ? 0 : 16;
  const float* p0 = A + (size_t)m * lda + kb;
  const float* p1 = p0 + 8;
  v16h r;
#pragma unroll
  for (int i = 0; i < 8; ++i) { r[i] = (_Float16)p0[i]; r[8 + i] = (_Float16)p1[i]; }
  return r;
}

// B fragment out of an LDS-staged 64x32 tile (row-major, 32 halves per row).
static __device__ inline v16h ld_b16_lds(const _Float16* sh, int j, int lane) {
  const int n  = lane & 15;
  const int kb = (lane < 16) ? 0 : 16;
  return *(const v16h*)(sh + (size_t)(16 * j + n) * 32 + kb);
}

static __device__ inline v8f wmma_f16(v16h a, v16h b, v8f c) {
  return __builtin_amdgcn_wmma_f32_16x16x32_f16(false, a, false, b, (short)0, c,
                                                false, false);
}

// ---------------------------------------------------------------------------
// Async global->LDS staging of one 64-row x 32-half B tile (4 KB).
// 256 threads, one GLOBAL_LOAD_ASYNC_TO_LDS_B128 each (ASYNCcnt-tracked).
// LDS offset = low 32 bits of the generic pointer (flat->LDS truncation).
// ---------------------------------------------------------------------------
static __device__ inline void stage_b_tile(_Float16* sh, const _Float16* g,
                                           int ldg, int tid) {
  const int row = tid >> 2;          // 64 rows
  const int part = tid & 3;          // 4 x 16B per row
  const uint32_t lds_off = (uint32_t)(uintptr_t)(void*)(sh + row * 32 + part * 8);
  const _Float16* gp = g + (size_t)row * ldg + part * 8;
  asm volatile("global_load_async_to_lds_b128 %0, %1, off"
               :: "v"(lds_off), "v"(gp) : "memory");
}

static __device__ inline void wait_async0() {
  asm volatile("s_wait_asynccnt 0" ::: "memory");
}

// ---------------------------------------------------------------------------
// LayerNorm over last dim (D_=1024) + convert to f16. One block per row.
// ---------------------------------------------------------------------------
__global__ __launch_bounds__(256) void ln_to_f16(const float* __restrict__ x,
                                                 const float* __restrict__ g,
                                                 const float* __restrict__ bta,
                                                 _Float16* __restrict__ y) {
  const int row = blockIdx.x;
  const int t = threadIdx.x;
  const float* xr = x + (size_t)row * D_;
  __shared__ float s1[256], s2[256];
  float a = 0.f, q = 0.f;
  for (int i = t; i < D_; i += 256) { float v = xr[i]; a += v; q += v * v; }
  s1[t] = a; s2[t] = q;
  __syncthreads();
  for (int s = 128; s > 0; s >>= 1) {
    if (t < s) { s1[t] += s1[t + s]; s2[t] += s2[t + s]; }
    __syncthreads();
  }
  const float mu  = s1[0] * (1.0f / D_);
  const float var = s2[0] * (1.0f / D_) - mu * mu;
  const float inv = rsqrtf(var + EPS_);
  for (int i = t; i < D_; i += 256) {
    float v = (xr[i] - mu) * inv * g[i] + bta[i];
    y[(size_t)row * D_ + i] = (_Float16)v;
  }
}

__global__ __launch_bounds__(256) void cvt_f32_f16(const float* __restrict__ x,
                                                   _Float16* __restrict__ y, int n) {
  int i = blockIdx.x * 256 + threadIdx.x;
  if (i < n) y[i] = (_Float16)x[i];
}

// ---------------------------------------------------------------------------
// C[M,N] = A[M,K] @ W[N,K]^T + bias ; 256-thread block = 8 waves, block tile
// 256x64, wave tile 32x64. B tile (64x32) async-staged to LDS, double buffered.
// MODE 0: f16 row-major ; MODE 1: f32 row-major ; MODE 2: f16 head-transposed
// (V projection -> Vt[(b*16+h)*64+d][t], ld=512).
// ---------------------------------------------------------------------------
template <int MODE>
__global__ __launch_bounds__(256) void gemm_xwT(const _Float16* __restrict__ A,
                                                const _Float16* __restrict__ W,
                                                const float* __restrict__ bias,
                                                void* __restrict__ out,
                                                int M, int N, int K) {
  __shared__ _Float16 Bsh[2][64 * 32];
  const int tid  = threadIdx.x;
  const int lane = tid & 31;
  const int wave = tid >> 5;
  const int m0 = blockIdx.x * 256 + wave * 32;
  const int n0 = blockIdx.y * 64;
  v8f acc[2][4] = {};

  stage_b_tile(Bsh[0], W + (size_t)n0 * K, K, tid);
  for (int k0 = 0; k0 < K; k0 += 32) {
    const int cur = (k0 >> 5) & 1;
    wait_async0();
    __syncthreads();
    if (k0 + 32 < K)
      stage_b_tile(Bsh[cur ^ 1], W + (size_t)n0 * K + (k0 + 32), K, tid);
    v16h af[2], bf[4];
#pragma unroll
    for (int i = 0; i < 2; ++i)
      af[i] = ld_a16(A + (size_t)(m0 + 16 * i) * K + k0, K, lane);
#pragma unroll
    for (int j = 0; j < 4; ++j)
      bf[j] = ld_b16_lds(Bsh[cur], j, lane);
#pragma unroll
    for (int i = 0; i < 2; ++i)
#pragma unroll
      for (int j = 0; j < 4; ++j)
        acc[i][j] = wmma_f16(af[i], bf[j], acc[i][j]);
  }

  const int nl = lane & 15;
  const int mb = (lane < 16) ? 0 : 8;
#pragma unroll
  for (int i = 0; i < 2; ++i) {
#pragma unroll
    for (int j = 0; j < 4; ++j) {
      const int col = n0 + 16 * j + nl;
      const float bv = bias[col];
#pragma unroll
      for (int r = 0; r < 8; ++r) {
        const int row = m0 + 16 * i + mb + r;
        const float v = acc[i][j][r] + bv;
        if (MODE == 0) {
          ((_Float16*)out)[(size_t)row * N + col] = (_Float16)v;
        } else if (MODE == 1) {
          ((float*)out)[(size_t)row * N + col] = v;
        } else {
          const int bb = row >> 9, tt = row & 511;
          const int hh = col >> 6, dd = col & 63;
          ((_Float16*)out)[(((size_t)(bb * 16 + hh) * 64 + dd) << 9) + tt] = (_Float16)v;
        }
      }
    }
  }
}

// ---------------------------------------------------------------------------
// scores[bh, q, k] = scale * Q[b,q,h*64:] . K[b,k,h*64:]  (masked -> NEG)
// grid (TQ/256, TK/64, B*H), block 256. K tile async-staged like the GEMM.
// ---------------------------------------------------------------------------
__global__ __launch_bounds__(256) void attn_scores(const _Float16* __restrict__ Q,
                                                   const _Float16* __restrict__ Kp,
                                                   const int* __restrict__ mask,
                                                   float* __restrict__ attn) {
  __shared__ _Float16 Bsh[2][64 * 32];
  const int tid  = threadIdx.x;
  const int lane = tid & 31;
  const int wave = tid >> 5;
  const int m0 = blockIdx.x * 256 + wave * 32;
  const int n0 = blockIdx.y * 64;
  const int bh = blockIdx.z;
  const int b = bh >> 4, h = bh & 15;
  const _Float16* Qb = Q + (size_t)b * TQ_ * D_ + h * HD_;
  const _Float16* Kb = Kp + (size_t)b * TK_ * D_ + h * HD_;
  v8f acc[2][4] = {};

  stage_b_tile(Bsh[0], Kb + (size_t)n0 * D_, D_, tid);
#pragma unroll
  for (int k0 = 0; k0 < HD_; k0 += 32) {
    const int cur = (k0 >> 5) & 1;
    wait_async0();
    __syncthreads();
    if (k0 + 32 < HD_)
      stage_b_tile(Bsh[cur ^ 1], Kb + (size_t)n0 * D_ + (k0 + 32), D_, tid);
    v16h af[2], bf[4];
#pragma unroll
    for (int i = 0; i < 2; ++i)
      af[i] = ld_a16(Qb + (size_t)(m0 + 16 * i) * D_ + k0, D_, lane);
#pragma unroll
    for (int j = 0; j < 4; ++j)
      bf[j] = ld_b16_lds(Bsh[cur], j, lane);
#pragma unroll
    for (int i = 0; i < 2; ++i)
#pragma unroll
      for (int j = 0; j < 4; ++j)
        acc[i][j] = wmma_f16(af[i], bf[j], acc[i][j]);
  }

  const int nl = lane & 15;
  const int mb = (lane < 16) ? 0 : 8;
  const float scale = 0.125f;  // 1/sqrt(64)
#pragma unroll
  for (int j = 0; j < 4; ++j) {
    const int col = n0 + 16 * j + nl;
    const bool valid = mask[b * TK_ + col] > 0;
#pragma unroll
    for (int i = 0; i < 2; ++i) {
#pragma unroll
      for (int r = 0; r < 8; ++r) {
        const int row = m0 + 16 * i + mb + r;
        attn[((size_t)bh * TQ_ + row) * TK_ + col] =
            valid ? acc[i][j][r] * scale : NEGV;
      }
    }
  }
}

// ---------------------------------------------------------------------------
// In-place row softmax over TK_=512 keys; zero rows if batch fully masked.
// ---------------------------------------------------------------------------
__global__ __launch_bounds__(256) void softmax_rows(float* __restrict__ attn,
                                                    const int* __restrict__ mask) {
  const int r = blockIdx.x;     // [0, B*H*TQ)
  const int b = r >> 13;        // H*TQ = 8192
  float* row = attn + (size_t)r * TK_;
  const int t = threadIdx.x;
  __shared__ float sA[256], sB[256];
  const float x0 = row[t], x1 = row[t + 256];
  const int vld = (mask[b * TK_ + t] > 0) | (mask[b * TK_ + t + 256] > 0);
  sA[t] = fmaxf(x0, x1);
  sB[t] = (float)vld;
  __syncthreads();
  for (int s = 128; s > 0; s >>= 1) {
    if (t < s) { sA[t] = fmaxf(sA[t], sA[t + s]); sB[t] += sB[t + s]; }
    __syncthreads();
  }
  const float mx = sA[0];
  const bool anyv = sB[0] > 0.f;
  __syncthreads();
  const float e0 = __expf(x0 - mx), e1 = __expf(x1 - mx);
  sA[t] = e0 + e1;
  __syncthreads();
  for (int s = 128; s > 0; s >>= 1) {
    if (t < s) sA[t] += sA[t + s];
    __syncthreads();
  }
  const float scl = anyv ? (1.0f / sA[0]) : 0.0f;
  row[t] = e0 * scl;
  row[t + 256] = e1 * scl;
}

// ---------------------------------------------------------------------------
// ctx[b, q, h*64+d] = sum_k attn[bh,q,k] * Vt[bh*64+d, k]   (f16 out)
// grid (TQ/256, 1, B*H), block 256; Vt tile async-staged to LDS.
// ---------------------------------------------------------------------------
__global__ __launch_bounds__(256) void attn_context(const float* __restrict__ attn,
                                                    const _Float16* __restrict__ Vt,
                                                    _Float16* __restrict__ ctx) {
  __shared__ _Float16 Bsh[2][64 * 32];
  const int tid  = threadIdx.x;
  const int lane = tid & 31;
  const int wave = tid >> 5;
  const int m0 = blockIdx.x * 256 + wave * 32;
  const int bh = blockIdx.z;
  const int b = bh >> 4, h = bh & 15;
  const float* Ab = attn + (size_t)bh * TQ_ * TK_;
  const _Float16* Bb = Vt + (size_t)bh * HD_ * TK_;
  v8f acc[2][4] = {};

  stage_b_tile(Bsh[0], Bb, TK_, tid);
  for (int k0 = 0; k0 < TK_; k0 += 32) {
    const int cur = (k0 >> 5) & 1;
    wait_async0();
    __syncthreads();
    if (k0 + 32 < TK_)
      stage_b_tile(Bsh[cur ^ 1], Bb + (k0 + 32), TK_, tid);
    v16h af[2], bf[4];
#pragma unroll
    for (int i = 0; i < 2; ++i)
      af[i] = ld_a32(Ab + (size_t)(m0 + 16 * i) * TK_ + k0, TK_, lane);
#pragma unroll
    for (int j = 0; j < 4; ++j)
      bf[j] = ld_b16_lds(Bsh[cur], j, lane);
#pragma unroll
    for (int i = 0; i < 2; ++i)
#pragma unroll
      for (int j = 0; j < 4; ++j)
        acc[i][j] = wmma_f16(af[i], bf[j], acc[i][j]);
  }

  const int nl = lane & 15;
  const int mb = (lane < 16) ? 0 : 8;
#pragma unroll
  for (int i = 0; i < 2; ++i) {
#pragma unroll
    for (int j = 0; j < 4; ++j) {
      const int col = 16 * j + nl;
#pragma unroll
      for (int r = 0; r < 8; ++r) {
        const int row = m0 + 16 * i + mb + r;
        ctx[((size_t)b * TQ_ + row) * D_ + h * HD_ + col] = (_Float16)acc[i][j][r];
      }
    }
  }
}

// ---------------------------------------------------------------------------
extern "C" void kernel_launch(void* const* d_in, const int* in_sizes, int n_in,
                              void* d_out, int out_size, void* d_ws, size_t ws_size,
                              hipStream_t stream) {
  const float* query = (const float*)d_in[0];
  const float* key_  = (const float*)d_in[1];
  const float* value = (const float*)d_in[2];
  const int*   mask  = (const int*)d_in[3];
  const float* q_ln_g = (const float*)d_in[4];
  const float* q_ln_b = (const float*)d_in[5];
  const float* k_ln_g = (const float*)d_in[6];
  const float* k_ln_b = (const float*)d_in[7];
  const float* v_ln_g = (const float*)d_in[8];
  const float* v_ln_b = (const float*)d_in[9];
  const float* Wq = (const float*)d_in[10];
  const float* bq = (const float*)d_in[11];
  const float* Wk = (const float*)d_in[12];
  const float* bk = (const float*)d_in[13];
  const float* Wv = (const float*)d_in[14];
  const float* bv = (const float*)d_in[15];
  const float* Wo = (const float*)d_in[16];
  const float* bo = (const float*)d_in[17];

  const size_t SZ_ACT = (size_t)B_ * TQ_ * D_;  // 8388608 elems
  const size_t SZ_W   = (size_t)D_ * D_;        // 1048576 elems

  _Float16* p = (_Float16*)d_ws;
  _Float16* qn   = p; p += SZ_ACT;
  _Float16* kn   = p; p += SZ_ACT;
  _Float16* vn   = p; p += SZ_ACT;
  _Float16* Wq16 = p; p += SZ_W;
  _Float16* Wk16 = p; p += SZ_W;
  _Float16* Wv16 = p; p += SZ_W;
  _Float16* Wo16 = p; p += SZ_W;
  _Float16* Qp   = p; p += SZ_ACT;
  _Float16* Kp   = p; p += SZ_ACT;
  _Float16* Vt   = p; p += SZ_ACT;
  _Float16* ctx  = p; p += SZ_ACT;

  float* out  = (float*)d_out;
  float* attn = out + SZ_ACT;  // (B,H,TQ,TK) fp32

  // 1) LayerNorm + f16 convert
  ln_to_f16<<<B_ * TQ_, 256, 0, stream>>>(query, q_ln_g, q_ln_b, qn);
  ln_to_f16<<<B_ * TK_, 256, 0, stream>>>(key_,  k_ln_g, k_ln_b, kn);
  ln_to_f16<<<B_ * TK_, 256, 0, stream>>>(value, v_ln_g, v_ln_b, vn);

  // 2) Weight converts
  cvt_f32_f16<<<(int)(SZ_W / 256), 256, 0, stream>>>(Wq, Wq16, (int)SZ_W);
  cvt_f32_f16<<<(int)(SZ_W / 256), 256, 0, stream>>>(Wk, Wk16, (int)SZ_W);
  cvt_f32_f16<<<(int)(SZ_W / 256), 256, 0, stream>>>(Wv, Wv16, (int)SZ_W);
  cvt_f32_f16<<<(int)(SZ_W / 256), 256, 0, stream>>>(Wo, Wo16, (int)SZ_W);

  // 3) Projections (M=8192, N=1024, K=1024)
  dim3 gg(8192 / 256, 1024 / 64);
  gemm_xwT<0><<<gg, 256, 0, stream>>>(qn, Wq16, bq, Qp, 8192, 1024, 1024);
  gemm_xwT<0><<<gg, 256, 0, stream>>>(kn, Wk16, bk, Kp, 8192, 1024, 1024);
  gemm_xwT<2><<<gg, 256, 0, stream>>>(vn, Wv16, bv, Vt, 8192, 1024, 1024);

  // 4) Scores + mask
  attn_scores<<<dim3(TQ_ / 256, TK_ / 64, B_ * H_), 256, 0, stream>>>(Qp, Kp, mask, attn);

  // 5) Softmax (in place in d_out attn region)
  softmax_rows<<<B_ * H_ * TQ_, 256, 0, stream>>>(attn, mask);

  // 6) Context
  attn_context<<<dim3(TQ_ / 256, 1, B_ * H_), 256, 0, stream>>>(attn, Vt, ctx);

  // 7) Output projection (f32 out)
  gemm_xwT<1><<<gg, 256, 0, stream>>>(ctx, Wo16, bo, out, 8192, 1024, 1024);
}
